// ff_net_24446953848957
// MI455X (gfx1250) — matrix operand order repaired and checked by reading, not compiled, not run on hardware
//
#include <hip/hip_runtime.h>
#include <hip/hip_bf16.h>

// CDNA5 (gfx1250) wave32 WMMA types
typedef __attribute__((ext_vector_type(16))) __bf16 v16bf;
typedef __attribute__((ext_vector_type(8)))  float  v8f;

#define NEG_SLOPE 0.1f

// ---------------------------------------------------------------------------
// Fragment loaders, contiguous k order k = 16*half + s for both A and B (not the CDNA5 ISA table order)
// A 16x32 bf16: lane = M, lanes 0-15: K=0..15, lanes 16-31: K=16..31
//   -> one contiguous 32B run per lane (merges to ds_load_b128)
// B 32x16 bf16: lane = N, lanes 0-15: K=0..15, lanes 16-31: K=16..31
//   -> one contiguous 32B run per lane (merges to 2x global_load_b128)
// ---------------------------------------------------------------------------
__device__ __forceinline__ v16bf load_a_frag(const __bf16* __restrict__ base,
                                             int ld, int lane) {
  const int half = (lane >> 4) & 1;
  const int r    = lane & 15;
  const __bf16* p = base + r * ld + (half << 4);
  v16bf f;
#pragma unroll
  for (int v = 0; v < 4; ++v) {
    f[2 * v]     = p[2 * v];
    f[2 * v + 1] = p[2 * v + 1];
    f[8 + 2 * v]     = p[8 + 2 * v];
    f[8 + 2 * v + 1] = p[8 + 2 * v + 1];
  }
  return f;
}

__device__ __forceinline__ v16bf load_b_frag(const __bf16* __restrict__ base,
                                             int ld, int lane) {
  const int half = (lane >> 4) & 1;
  const int c    = lane & 15;
  const __bf16* p = base + c * ld + (half << 4);
  v16bf f;
#pragma unroll
  for (int v = 0; v < 16; ++v) f[v] = p[v];
  return f;
}

template <int ACT>
__device__ __forceinline__ float act_apply(float v) {
  if (ACT == 1) return __builtin_amdgcn_rcpf(1.0f + __expf(-v)); // fast sigmoid
  if (ACT == 2) return (v > 0.0f) ? v : NEG_SLOPE * v;           // leaky relu
  return v;
}

// ---------------------------------------------------------------------------
// Fused layer on a 32-row LDS-resident tile:
//   Y[32,NN] = act( X[32,KP](LDS bf16) * W[NN,KP](global bf16)^T + bias )
// K-outer / N-tile-inner with persistent accumulators: each wave keeps up to
// 4 N-tiles (x2 M-tiles) of accumulators in VGPRs, loads each A fragment ONCE
// per K-step and streams 4 B fragments + 8 WMMAs against it (A DS traffic /4,
// B L2 traffic /2 vs naive). Scalar (readfirstlane) loop control keeps EXEC
// all-ones around every WMMA.
// ---------------------------------------------------------------------------
template <int KP, int NN, int ACT, bool OB, bool OF>
__device__ __forceinline__ void gemm_tile32(
    const __bf16* __restrict__ Xin, int ldx,
    const __bf16* __restrict__ W, const float* __restrict__ bias,
    __bf16* outB, int ldob, float* outF, int ldof) {
  const int lane = threadIdx.x & 31;
  const int wave = __builtin_amdgcn_readfirstlane(threadIdx.x >> 5);
  constexpr int nTiles = NN >> 4;
  // N-tiles per wave per group (register blocking factor, capped at 4)
  constexpr int NTI    = (nTiles >= 32) ? 4 : ((nTiles >= 8) ? (nTiles / 8) : 1);
  constexpr int GROUPS = (nTiles + 8 * NTI - 1) / (8 * NTI);
  constexpr bool GUARD = (nTiles < 8);   // some waves idle on tiny layers

  __builtin_prefetch(W, 0, 1); // global_prefetch_b8 on the weight panel

#pragma unroll
  for (int g = 0; g < GROUPS; ++g) {
    const int nt0 = wave + g * 8 * NTI;       // wave's tiles: nt0 + 8*i
    if (!GUARD || nt0 < nTiles) {             // wave-uniform scalar branch
      v8f acc0[NTI];
      v8f acc1[NTI];
#pragma unroll
      for (int i = 0; i < NTI; ++i) { acc0[i] = {}; acc1[i] = {}; }

      for (int ks = 0; ks < KP; ks += 32) {
        v16bf a0 = load_a_frag(Xin + ks, ldx, lane);
        v16bf a1 = load_a_frag(Xin + (ldx << 4) + ks, ldx, lane);
#pragma unroll
        for (int i = 0; i < NTI; ++i) {
          const int nt = nt0 + 8 * i;
          v16bf b = load_b_frag(W + (nt << 4) * KP + ks, KP, lane);
          acc0[i] = __builtin_amdgcn_wmma_f32_16x16x32_bf16(
              false, a0, false, b, (short)0, acc0[i], false, false);
          acc1[i] = __builtin_amdgcn_wmma_f32_16x16x32_bf16(
              false, a1, false, b, (short)0, acc1[i], false, false);
        }
      }

      const int half = (lane >> 4) & 1;
#pragma unroll
      for (int i = 0; i < NTI; ++i) {
        const int nt = nt0 + 8 * i;
        const int nc = (nt << 4) + (lane & 15);
        const float bv = bias[nc];
#pragma unroll
        for (int r = 0; r < 8; ++r) {
          // D layout: VGPR r -> M=r (lanes 0-15) / M=8+r (lanes 16-31), N=lane
          const int m  = (half << 3) + r;
          float v0 = act_apply<ACT>(acc0[i][r] + bv);
          float v1 = act_apply<ACT>(acc1[i][r] + bv);
          if (OB) {
            outB[m * ldob + nc]        = (__bf16)v0;
            outB[(16 + m) * ldob + nc] = (__bf16)v1;
          }
          if (OF) {
            outF[m * ldof + nc]        = v0;
            outF[(16 + m) * ldof + nc] = v1;
          }
        }
      }
    }
  }
}

// ---------------------------------------------------------------------------
// Weight f32 -> bf16 with zero K-padding (K padded to multiple of 32)
// ---------------------------------------------------------------------------
__global__ void cvt_pad_kernel(const float* __restrict__ src, __bf16* __restrict__ dst,
                               int dout, int din, int dinPad) {
  int i = blockIdx.x * blockDim.x + threadIdx.x;
  int total = dout * dinPad;
  if (i >= total) return;
  int r = i / dinPad;
  int k = i - r * dinPad;
  float v = (k < din) ? src[r * din + k] : 0.0f;
  dst[i] = (__bf16)v;
}

// ---------------------------------------------------------------------------
// er_net: 32 rows per workgroup, fully fused chain 64->1024->512->128->32->64
// ---------------------------------------------------------------------------
struct ErArgs {
  const float* egoY;
  const float* selW;    const float* selB;
  const float* sin0W;   const float* sin0B;
  const float* ensin0W; const float* ensin0B;
  const __bf16* Wl1;  const float* bl1;
  const __bf16* Wl2;  const float* bl2;
  const __bf16* Wl3;  const float* bl3;
  const __bf16* Wl4;  const float* bl4;
  const __bf16* Wefc; const float* befc;
  float* out;
};

__global__ __launch_bounds__(256) void er_kernel(ErArgs A) {
  extern __shared__ char smem[];
  __bf16* bufA = (__bf16*)smem;                 // 32x1024 bf16 = 64 KB
  __bf16* bufB = (__bf16*)(smem + 65536);       // 32x1024 bf16 = 64 KB
  float*  sinT = (float*)(smem + 131072);       // 32x32 f32
  __bf16* X0   = bufB;                          // 32x64 bf16 (aliases bufB)

  const int tid  = threadIdx.x;
  const int row  = tid & 31;
  const int part = tid >> 5;                    // 8 segments per row
  const int rb   = blockIdx.x * 32;

  // ---- feature build (scalar, f32) ----
  const float LM[4] = {13.55f, 17.45f, 21.12f, 24.91f};
  float y = A.egoY[rb + row];
  float d[4], x[4], z[4];
  float mx = -1e30f;
#pragma unroll
  for (int j = 0; j < 4; ++j) d[j] = LM[j] - y;
#pragma unroll
  for (int i = 0; i < 4; ++i) {
    float s = A.selB[i];
#pragma unroll
    for (int j = 0; j < 4; ++j) s += A.selW[i * 4 + j] * d[j];
    z[i] = s;
    mx = fmaxf(mx, s);
  }
  float se = 0.0f;
#pragma unroll
  for (int i = 0; i < 4; ++i) { z[i] = __expf(z[i] - mx); se += z[i]; }
  float inv_se = __builtin_amdgcn_rcpf(se);
#pragma unroll
  for (int i = 0; i < 4; ++i) x[i] = z[i] * inv_se * d[i];

  // sin0: each of 8 parts computes 4 of 32 outputs
#pragma unroll
  for (int j = 0; j < 4; ++j) {
    int o = part * 4 + j;
    float s = A.sin0B[o];
#pragma unroll
    for (int k = 0; k < 4; ++k) s += A.sin0W[o * 4 + k] * x[k];
    sinT[row * 32 + o] = __sinf(s);
  }
  __syncthreads();

  // ensin0 (32x32) + assemble 48-feature row (padded to 64) as bf16
#pragma unroll
  for (int j = 0; j < 4; ++j) {
    int o = part * 4 + j;
    float e = A.ensin0B[o];
    for (int k = 0; k < 32; ++k) e += A.ensin0W[o * 32 + k] * sinT[row * 32 + k];
    X0[row * 64 + 4 + o] = (__bf16)e;
  }
  if (part == 0) for (int j = 0; j < 4; ++j) X0[row * 64 + j]      = (__bf16)x[j];
  if (part == 1) for (int j = 0; j < 4; ++j) X0[row * 64 + 36 + j] = (__bf16)(x[j] * x[j]);
  if (part == 2) for (int j = 0; j < 4; ++j) X0[row * 64 + 40 + j] = (__bf16)(x[j] * x[j] * x[j]);
  if (part == 3) for (int j = 0; j < 4; ++j) X0[row * 64 + 44 + j] = (__bf16)__expf(x[j]);
  if (part == 4) for (int j = 0; j < 16; ++j) X0[row * 64 + 48 + j] = (__bf16)0.0f;
  __syncthreads();

  // ---- fused WMMA chain ----
  gemm_tile32<64, 1024, 1, true, false>(X0,   64,   A.Wl1, A.bl1, bufA, 1024, nullptr, 0);
  __syncthreads();
  gemm_tile32<1024, 512, 2, true, false>(bufA, 1024, A.Wl2, A.bl2, bufB, 512, nullptr, 0);
  __syncthreads();
  gemm_tile32<512, 128, 2, true, false>(bufB, 512,  A.Wl3, A.bl3, bufA, 128, nullptr, 0);
  __syncthreads();
  gemm_tile32<128, 32, 2, true, false>(bufA, 128,  A.Wl4, A.bl4, bufB, 32,  nullptr, 0);
  __syncthreads();
  gemm_tile32<32, 64, 0, false, true>(bufB, 32, A.Wefc, A.befc, nullptr, 0,
                                      A.out + (size_t)rb * 64, 64);   // -> Er
}

// ---------------------------------------------------------------------------
// en_net: one workgroup == one batch b (32 neighbor rows), fully fused
// including neighbor softmax + weighted sum.
// ---------------------------------------------------------------------------
struct EnArgs {
  const float *Pn, *Pego, *Vn, *Vego, *Cn;
  const float* sin1W; const float* sin1b;
  const __bf16* Wensin1; const float* bensin1;
  const __bf16* Wm1; const float* bm1;
  const __bf16* Wm2; const float* bm2;
  const __bf16* Wa;  const float* ba;
  const __bf16* Wb_; const float* bb_;
  const __bf16* Wc;  const float* bc;
  const __bf16* Ww1; const float* bw1;
  const __bf16* Ww2; const float* bw2;
  const float* w3W; const float* w3b;
  float* out;
};

__global__ __launch_bounds__(256) void en_kernel(EnArgs A) {
  extern __shared__ char smem[];
  __bf16* bufA  = (__bf16*)(smem);                // 32x512 bf16  (32 KB)
  __bf16* bufB  = (__bf16*)(smem + 32768);        // 32x512 bf16  (32 KB)
  __bf16* ninfo = (__bf16*)(smem + 65536);        // 32x288 bf16  (18 KB)
  __bf16* sinP  = (__bf16*)(smem + 83968);        // 32x128 bf16  (8 KB)
  __bf16* sinV  = (__bf16*)(smem + 92160);        // 32x128 bf16  (8 KB)
  __bf16* sbuf  = (__bf16*)(smem + 100352);       // 32x96 bf16   (6 KB)
  float*  EnF32 = (float*)(smem + 106496);        // 32x64 f32    (8 KB)
  float*  sval  = (float*)(smem + 114688);        // 32 f32
  float*  wsm   = (float*)(smem + 114816);        // 32 f32

  const int tid  = threadIdx.x;
  const int row  = tid & 31;                      // neighbor index n
  const int part = tid >> 5;
  const int b    = blockIdx.x;
  const int ri   = b * 32 + row;

  // ---- phase 1: deltas + sin inputs (scalar f32) ----
  const float pex = A.Pego[b * 2 + 0], pey = A.Pego[b * 2 + 1];
  const float vex = A.Vego[b * 2 + 0], vey = A.Vego[b * 2 + 1];
  const float pnx = A.Pn[ri * 2 + 0],  pny = A.Pn[ri * 2 + 1];
  const float vnx = A.Vn[ri * 2 + 0],  vny = A.Vn[ri * 2 + 1];
  const float dpx = pnx - pex, dpy = pny - pey;
  const float dvx = vnx - vex, dvy = vny - vey;

#pragma unroll 4
  for (int j = 0; j < 16; ++j) {
    int o = part * 16 + j;
    float w0 = A.sin1W[o * 2 + 0], w1 = A.sin1W[o * 2 + 1], sb0 = A.sin1b[o];
    sinP[row * 128 + o] = (__bf16)__sinf(w0 * dpx + w1 * dpy + sb0);
    sinV[row * 128 + o] = (__bf16)__sinf(w0 * dvx + w1 * dvy + sb0);
  }
  if (part == 0) {
    const float cx = A.Cn[ri * 2 + 0], cy = A.Cn[ri * 2 + 1];
    __bf16* nf = ninfo + row * 288;
    nf[0] = (__bf16)pex; nf[1] = (__bf16)pey; nf[2] = (__bf16)pnx; nf[3] = (__bf16)pny;
    nf[4] = (__bf16)dpx; nf[5] = (__bf16)dpy; nf[6] = (__bf16)dpx; nf[7] = (__bf16)dpy;
    nf[136] = (__bf16)(dpx * dpx);       nf[137] = (__bf16)(dpy * dpy);
    nf[138] = (__bf16)(dpx * dpx * dpx); nf[139] = (__bf16)(dpy * dpy * dpy);
    nf[140] = (__bf16)__expf(dpx);       nf[141] = (__bf16)__expf(dpy);
    nf[142] = (__bf16)vex; nf[143] = (__bf16)vey; nf[144] = (__bf16)vnx; nf[145] = (__bf16)vny;
    nf[146] = (__bf16)dvx; nf[147] = (__bf16)dvy; nf[148] = (__bf16)dvx; nf[149] = (__bf16)dvy;
    nf[278] = (__bf16)(dvx * dvx);       nf[279] = (__bf16)(dvy * dvy);
    nf[280] = (__bf16)(dvx * dvx * dvx); nf[281] = (__bf16)(dvy * dvy * dvy);
    nf[282] = (__bf16)__expf(dvx);       nf[283] = (__bf16)__expf(dvy);
    nf[284] = (__bf16)cx; nf[285] = (__bf16)cy;
    nf[286] = (__bf16)0.0f; nf[287] = (__bf16)0.0f;
    __bf16* sb = sbuf + row * 96;
    sb[64] = (__bf16)pnx; sb[65] = (__bf16)pny; sb[66] = (__bf16)dpx; sb[67] = (__bf16)dpy;
  }
  if (part == 1) {
    __bf16* sb = sbuf + row * 96;
    for (int j = 68; j < 96; ++j) sb[j] = (__bf16)0.0f;
  }
  __syncthreads();

  // ensin1 GEMMs write straight into Ninfo feature slots (disjoint columns)
  gemm_tile32<128, 128, 0, true, false>(sinP, 128, A.Wensin1, A.bensin1,
                                        ninfo + 8, 288, nullptr, 0);
  gemm_tile32<128, 128, 0, true, false>(sinV, 128, A.Wensin1, A.bensin1,
                                        ninfo + 150, 288, nullptr, 0);
  __syncthreads();

  // ---- main chain ----
  gemm_tile32<288, 512, 2, true, false>(ninfo, 288, A.Wm1, A.bm1, bufA, 512, nullptr, 0);
  __syncthreads();
  gemm_tile32<512, 512, 0, true, false>(bufA, 512, A.Wm2, A.bm2, bufB, 512, nullptr, 0);
  __syncthreads();
  gemm_tile32<512, 512, 1, true, false>(bufB, 512, A.Wa, A.ba, bufA, 512, nullptr, 0);
  __syncthreads();
  gemm_tile32<512, 256, 2, true, false>(bufA, 512, A.Wb_, A.bb_, bufB, 256, nullptr, 0);
  __syncthreads();
  // En: bf16 into scoring concat buffer AND f32 copy for the final reduction
  gemm_tile32<256, 64, 0, true, true>(bufB, 256, A.Wc, A.bc, sbuf, 96, EnF32, 64);
  __syncthreads();

  // ---- scoring head ----
  gemm_tile32<96, 256, 1, true, false>(sbuf, 96, A.Ww1, A.bw1, bufA, 256, nullptr, 0);
  __syncthreads();
  gemm_tile32<256, 128, 2, true, false>(bufA, 256, A.Ww2, A.bw2, bufB, 128, nullptr, 0);
  __syncthreads();

  if (tid < 32) {
    float s = A.w3b[0];
    for (int k = 0; k < 128; ++k) s += (float)bufB[tid * 128 + k] * A.w3W[k];
    sval[tid] = s;
  }
  __syncthreads();
  if (tid == 0) {
    float m = sval[0];
    for (int n = 1; n < 32; ++n) m = fmaxf(m, sval[n]);
    float ss = 0.0f;
    for (int n = 0; n < 32; ++n) { float e = __expf(sval[n] - m); wsm[n] = e; ss += e; }
    float inv = __builtin_amdgcn_rcpf(ss);
    for (int n = 0; n < 32; ++n) wsm[n] *= inv;
  }
  __syncthreads();
  if (tid < 64) {
    float acc = 0.0f;
    for (int n = 0; n < 32; ++n) acc += wsm[n] * EnF32[n * 64 + tid];
    A.out[(size_t)2048 * 64 + (size_t)b * 64 + tid] = acc;   // En block
  }
}

// ---------------------------------------------------------------------------
extern "C" void kernel_launch(void* const* d_in, const int* in_sizes, int n_in,
                              void* d_out, int out_size, void* d_ws, size_t ws_size,
                              hipStream_t stream) {
  (void)in_sizes; (void)n_in; (void)out_size; (void)ws_size;

  // d_in order = setup_inputs() dict order; params tuples flatten W then b.
  const float* Pn    = (const float*)d_in[0];
  const float* Pego  = (const float*)d_in[1];
  const float* Vn    = (const float*)d_in[2];
  const float* Vego  = (const float*)d_in[3];
  const float* Cn    = (const float*)d_in[4];
  const float* egoY  = (const float*)d_in[5];
  const float* selW  = (const float*)d_in[6];
  const float* selB  = (const float*)d_in[7];
  const float* l1W   = (const float*)d_in[8];   const float* l1b  = (const float*)d_in[9];
  const float* l2W   = (const float*)d_in[10];  const float* l2b  = (const float*)d_in[11];
  const float* l3W   = (const float*)d_in[12];  const float* l3b  = (const float*)d_in[13];
  const float* l4W   = (const float*)d_in[14];  const float* l4b  = (const float*)d_in[15];
  const float* efcW  = (const float*)d_in[16];  const float* efcb = (const float*)d_in[17];
  const float* sin0W = (const float*)d_in[18];  const float* sin0b = (const float*)d_in[19];
  const float* ens0W = (const float*)d_in[20];  const float* ens0b = (const float*)d_in[21];
  const float* sin1W = (const float*)d_in[22];  const float* sin1b = (const float*)d_in[23];
  const float* ens1W = (const float*)d_in[24];  const float* ens1b = (const float*)d_in[25];
  const float* m1W   = (const float*)d_in[26];  const float* m1b  = (const float*)d_in[27];
  const float* m2W   = (const float*)d_in[28];  const float* m2b  = (const float*)d_in[29];
  const float* aW    = (const float*)d_in[30];  const float* ab   = (const float*)d_in[31];
  const float* bW    = (const float*)d_in[32];  const float* bb   = (const float*)d_in[33];
  const float* cW    = (const float*)d_in[34];  const float* cb   = (const float*)d_in[35];
  const float* w1W   = (const float*)d_in[36];  const float* w1b  = (const float*)d_in[37];
  const float* w2W   = (const float*)d_in[38];  const float* w2b  = (const float*)d_in[39];
  const float* w3W   = (const float*)d_in[40];  const float* w3b  = (const float*)d_in[41];

  float* out = (float*)d_out;

  // bf16 weight arena in workspace (K padded to multiples of 32, zero-filled)
  char* ws = (char*)d_ws;
  size_t ofs = 0;
  auto alloc = [&](size_t elems) {
    __bf16* p = (__bf16*)(ws + ofs);
    ofs += ((elems * 2 + 255) / 256) * 256;   // 256B align each matrix
    return p;
  };
  __bf16* W_l1  = alloc((size_t)1024 * 64);
  __bf16* W_l2  = alloc((size_t)512 * 1024);
  __bf16* W_l3  = alloc((size_t)128 * 512);
  __bf16* W_l4  = alloc((size_t)32 * 128);
  __bf16* W_efc = alloc((size_t)64 * 32);
  __bf16* W_es1 = alloc((size_t)128 * 128);
  __bf16* W_m1  = alloc((size_t)512 * 288);
  __bf16* W_m2  = alloc((size_t)512 * 512);
  __bf16* W_a   = alloc((size_t)512 * 512);
  __bf16* W_b   = alloc((size_t)256 * 512);
  __bf16* W_c   = alloc((size_t)64 * 256);
  __bf16* W_w1  = alloc((size_t)256 * 96);
  __bf16* W_w2  = alloc((size_t)128 * 256);

  auto cvt = [&](const float* src, __bf16* dst, int dout, int din, int dinPad) {
    int total = dout * dinPad;
    cvt_pad_kernel<<<(total + 255) / 256, 256, 0, stream>>>(src, dst, dout, din, dinPad);
  };
  cvt(l1W,  W_l1,  1024, 48,  64);
  cvt(l2W,  W_l2,  512, 1024, 1024);
  cvt(l3W,  W_l3,  128, 512,  512);
  cvt(l4W,  W_l4,  32,  128,  128);
  cvt(efcW, W_efc, 64,  32,   32);
  cvt(ens1W, W_es1, 128, 128, 128);
  cvt(m1W,  W_m1,  512, 286,  288);
  cvt(m2W,  W_m2,  512, 512,  512);
  cvt(aW,   W_a,   512, 512,  512);
  cvt(bW,   W_b,   256, 512,  512);
  cvt(cW,   W_c,   64,  256,  256);
  cvt(w1W,  W_w1,  256, 68,   96);
  cvt(w2W,  W_w2,  128, 256,  256);

  // er branch: 2048 rows / 32 per WG
  ErArgs ea;
  ea.egoY = egoY;
  ea.selW = selW;   ea.selB = selB;
  ea.sin0W = sin0W; ea.sin0B = sin0b;
  ea.ensin0W = ens0W; ea.ensin0B = ens0b;
  ea.Wl1 = W_l1; ea.bl1 = l1b;
  ea.Wl2 = W_l2; ea.bl2 = l2b;
  ea.Wl3 = W_l3; ea.bl3 = l3b;
  ea.Wl4 = W_l4; ea.bl4 = l4b;
  ea.Wefc = W_efc; ea.befc = efcb;
  ea.out = out;
  const size_t ER_SMEM = 65536 + 65536 + 4096;            // 135168 B
  er_kernel<<<64, 256, ER_SMEM, stream>>>(ea);

  // en branch: 2048 batches, one WG each (32 neighbors per WG)
  EnArgs na;
  na.Pn = Pn; na.Pego = Pego; na.Vn = Vn; na.Vego = Vego; na.Cn = Cn;
  na.sin1W = sin1W; na.sin1b = sin1b;
  na.Wensin1 = W_es1; na.bensin1 = ens1b;
  na.Wm1 = W_m1; na.bm1 = m1b;
  na.Wm2 = W_m2; na.bm2 = m2b;
  na.Wa  = W_a;  na.ba  = ab;
  na.Wb_ = W_b;  na.bb_ = bb;
  na.Wc  = W_c;  na.bc  = cb;
  na.Ww1 = W_w1; na.bw1 = w1b;
  na.Ww2 = W_w2; na.bw2 = w2b;
  na.w3W = w3W;  na.w3b = w3b;
  na.out = out;
  const size_t EN_SMEM = 114944;                          // bufs + reductions
  en_kernel<<<2048, 256, EN_SMEM, stream>>>(na);
}
